// GD_Attention_29755533426967
// MI455X (gfx1250) — matrix operand order repaired and checked by reading, not compiled, not run on hardware
//
#include <hip/hip_runtime.h>
#include <hip/hip_bf16.h>

// ---------------------------------------------------------------------------
// LeViT attention, fused, bf16 WMMA on gfx1250 (wave32).
//   B=8, C=384, N=1024, heads=8, key_dim=16, d_v=64
// ---------------------------------------------------------------------------

#define B_    8
#define C_IN  384
#define N_    1024
#define NH    8
#define KD    16
#define DV    64
#define DH    (NH * DV)   // 512
#define OQKV  768         // 128 q + 128 k + 512 v

typedef __attribute__((ext_vector_type(16))) __bf16 v16bf;
typedef __attribute__((ext_vector_type(8)))  __bf16 v8bf;
typedef __attribute__((ext_vector_type(8)))  float  v8f;

__device__ __forceinline__ __bf16 to_bf16(float f) {
  unsigned int u = __builtin_bit_cast(unsigned int, f);
  u += 0x7FFFu + ((u >> 16) & 1u);            // round-to-nearest-even
  unsigned short s = (unsigned short)(u >> 16);
  return __builtin_bit_cast(__bf16, s);
}

__device__ __forceinline__ v8bf ld8(const __bf16* p) {
  return *(const v8bf*)p;                      // 16B global_load_b128
}

__device__ __forceinline__ v16bf cat16(v8bf lo, v8bf hi) {
  return __builtin_shufflevector(lo, hi, 0,1,2,3,4,5,6,7,8,9,10,11,12,13,14,15);
}

__device__ __forceinline__ v8f wmma_bf16(v16bf a, v16bf b, v8f c) {
  return __builtin_amdgcn_wmma_f32_16x16x32_bf16(false, a, false, b, (short)0, c,
                                                 false, false);
}

// ---------------------------------------------------------------------------
// Kernel 1: pack x -> xT bf16 [b][n][384]; weights -> bf16 (qkv fused, wp)
// ---------------------------------------------------------------------------
__global__ void prep_pack(const float* __restrict__ x,
                          const float* __restrict__ wq, const float* __restrict__ wk,
                          const float* __restrict__ wv, const float* __restrict__ wp,
                          __bf16* __restrict__ xT, __bf16* __restrict__ wqkv,
                          __bf16* __restrict__ wpb) {
  long i = (long)blockIdx.x * blockDim.x + threadIdx.x;
  const long NX = (long)B_ * N_ * C_IN;      // 3145728
  const long NW = (long)OQKV * C_IN;         // 294912
  const long NP = (long)C_IN * DH;           // 196608
  if (i < NX) {
    int c = (int)(i % C_IN);
    long bn = i / C_IN;
    int n = (int)(bn % N_);
    int b = (int)(bn / N_);
    xT[i] = to_bf16(x[((long)b * C_IN + c) * N_ + n]);
  } else if (i < NX + NW) {
    long j = i - NX;
    int o = (int)(j / C_IN), c = (int)(j % C_IN);
    float f = (o < 128) ? wq[o * C_IN + c]
            : (o < 256) ? wk[(o - 128) * C_IN + c]
                        : wv[(o - 256) * C_IN + c];
    wqkv[j] = to_bf16(f);
  } else if (i < NX + NW + NP) {
    long j = i - NX - NW;
    wpb[j] = to_bf16(wp[j]);
  }
}

// ---------------------------------------------------------------------------
// Kernel 2: fused QKV projection GEMM + folded BN.
//   One wave computes a 32(o) x 64(n) tile: 8 WMMA per 32-wide K step,
//   12 fragment loads (2 A + 4 B) per step. All fragments of a K-step are
//   loaded before the WMMA chain so loads overlap compute.
//   q,k -> [b][h][n][16]  (row-per-query layout for attention fragments)
//   v   -> [b][h][d][n]   (row-per-d layout for the P*V^T B-fragment)
// ---------------------------------------------------------------------------
__global__ void qkv_gemm(const __bf16* __restrict__ wqkv, const __bf16* __restrict__ xT,
                         const float* __restrict__ sq, const float* __restrict__ bq,
                         const float* __restrict__ sk, const float* __restrict__ bk,
                         const float* __restrict__ sv, const float* __restrict__ bv,
                         __bf16* __restrict__ qo, __bf16* __restrict__ ko,
                         __bf16* __restrict__ vo) {
  int gw   = (int)((blockIdx.x * blockDim.x + threadIdx.x) >> 5);
  int lane = threadIdx.x & 31;
  int lx   = lane & 15;
  bool hi  = lane >= 16;
  // waves: b(8) * o_group(24, 32 channels each) * n_group(16, 64 cols each)
  int ng = gw & 15;
  int rest = gw >> 4;
  int og = rest % 24;
  int b  = rest / 24;
  int o0 = og * 32;
  int nbase = ng * 64;

  const __bf16* wrow0 = wqkv + (long)(o0 + lx) * C_IN;
  const __bf16* wrow1 = wqkv + (long)(o0 + 16 + lx) * C_IN;
  const __bf16* xrow  = xT + ((long)b * N_ + nbase + lx) * C_IN;

  v8f acc[2][4] = {{{}, {}, {}, {}}, {{}, {}, {}, {}}};
  for (int c0 = 0; c0 < C_IN; c0 += 32) {
    int aoff = c0 + (hi ? 8 : 0);
    int boff = c0 + (hi ? 8 : 0);
    // issue all loads of this K-step first, then the WMMA chain
    v16bf af0 = cat16(ld8(wrow0 + aoff), ld8(wrow0 + aoff + 16));
    v16bf af1 = cat16(ld8(wrow1 + aoff), ld8(wrow1 + aoff + 16));
    v16bf bfrag[4];
#pragma unroll
    for (int t = 0; t < 4; ++t) {
      const __bf16* xr = xrow + (long)t * 16 * C_IN;
      bfrag[t] = cat16(ld8(xr + boff), ld8(xr + boff + 16));
    }
#pragma unroll
    for (int t = 0; t < 4; ++t) {
      acc[0][t] = wmma_bf16(af0, bfrag[t], acc[0][t]);
      acc[1][t] = wmma_bf16(af1, bfrag[t], acc[1][t]);
    }
  }

#pragma unroll
  for (int u = 0; u < 2; ++u) {
#pragma unroll
    for (int t = 0; t < 4; ++t) {
      int n = nbase + t * 16 + lx;
#pragma unroll
      for (int r = 0; r < 8; ++r) {
        int o = o0 + u * 16 + r + (hi ? 8 : 0);
        float val = acc[u][t][r];
        if (o < 128) {
          val = val * sq[o] + bq[o];
          int h = o >> 4, c = o & 15;
          qo[(((long)b * NH + h) * N_ + n) * KD + c] = to_bf16(val);
        } else if (o < 256) {
          int oo = o - 128;
          val = val * sk[oo] + bk[oo];
          int h = oo >> 4, c = oo & 15;
          ko[(((long)b * NH + h) * N_ + n) * KD + c] = to_bf16(val);
        } else {
          int oo = o - 256;
          val = val * sv[oo] + bv[oo];
          int h = oo >> 6, d = oo & 63;
          vo[(((long)b * NH + h) * DV + d) * N_ + n] = to_bf16(val);
        }
      }
    }
  }
}

// ---------------------------------------------------------------------------
// Kernel 3: fused flash attention per (b,h,16-query tile).
//   S^T = K^T * Q  so the softmax'd P lands directly in the A-fragment
//   layout of the P*V^T WMMA (no cross-lane shuffles for P).
//   V fragments are loaded before the softmax VALU so they overlap it.
//   Online softmax over 1024 keys in blocks of 32; ReLU fused on store.
//   Output xx bf16 [b][n][512] (c = h*64+d), ready as proj B-fragment rows.
// ---------------------------------------------------------------------------
__global__ void attn_fused(const __bf16* __restrict__ qb, const __bf16* __restrict__ kb,
                           const __bf16* __restrict__ vb, __bf16* __restrict__ xx) {
  int gw   = (int)((blockIdx.x * blockDim.x + threadIdx.x) >> 5);
  int lane = threadIdx.x & 31;
  int lx   = lane & 15;
  bool hiL = lane >= 16;
  int nt = gw & 63;
  int bh = gw >> 6;           // b*8+h
  int n0 = nt * 16;
  int h = bh & 7, b = bh >> 3;

  // Q B-fragment (K dim = channel c, padded 16->32): elements 0-7 hold c = 8*half + s, elements 8-15 are zero pad.
  v16bf qf = {};
  {
    const __bf16* qrow = qb + ((long)bh * N_ + n0 + lx) * KD;
    qf = cat16(ld8(qrow + (hiL ? 8 : 0)), z8);
  }

  const __bf16* kbase = kb + (long)bh * N_ * KD;
  const __bf16* vbase = vb + (long)bh * DV * N_;

  float runM = -3.0e38f;
  float runS = 0.0f;
  v8f oacc[4] = {{}, {}, {}, {}};
  v8f czero = {};
  v8bf z8 = {};

  for (int m0 = 0; m0 < N_; m0 += 32) {
    // ---- issue ALL loads of this block first (K rows + V rows) ----
    int coff = hiL ? 8 : 0;
    v16bf ak0 = cat16(ld8(kbase + (long)(m0 + lx) * KD + coff), z8);
    v16bf ak1 = cat16(ld8(kbase + (long)(m0 + 16 + lx) * KD + coff), z8);
    int mcol = m0 + (hiL ? 8 : 0);
    v16bf vf[4];
#pragma unroll
    for (int t = 0; t < 4; ++t) {
      const __bf16* vrow = vbase + (long)(t * 16 + lx) * N_ + mcol;
      vf[t] = cat16(ld8(vrow), ld8(vrow + 16));
    }

    // ---- S^T tiles ----
    v8f s0 = wmma_bf16(ak0, qf, czero);
    v8f s1 = wmma_bf16(ak1, qf, czero);

    // ---- online softmax: lane holds n = lx, 16 of 32 m values;
    //      row is split across lane pair (j, j+16) -> one xor-16 reduce.
    float mx = s0[0];
#pragma unroll
    for (int r = 1; r < 8; ++r) mx = fmaxf(mx, s0[r]);
#pragma unroll
    for (int r = 0; r < 8; ++r) mx = fmaxf(mx, s1[r]);
    mx = fmaxf(mx, __shfl_xor(mx, 16, 32));
    float newM = fmaxf(runM, mx);
    float alpha = __expf(runM - newM);
    float p0[8], p1[8];
    float lsum = 0.0f;
#pragma unroll
    for (int r = 0; r < 8; ++r) { p0[r] = __expf(s0[r] - newM); lsum += p0[r]; }
#pragma unroll
    for (int r = 0; r < 8; ++r) { p1[r] = __expf(s1[r] - newM); lsum += p1[r]; }
    lsum += __shfl_xor(lsum, 16, 32);
    runS = runS * alpha + lsum;
    runM = newM;

    // rescale accumulators: reg r belongs to query n_rel = r + (hiL?8:0);
    // alpha for that row lives (identically) on lanes n_rel and n_rel+16.
    float alr[8];
#pragma unroll
    for (int r = 0; r < 8; ++r) alr[r] = __shfl(alpha, r + (hiL ? 8 : 0), 32);
#pragma unroll
    for (int t = 0; t < 4; ++t)
#pragma unroll
      for (int r = 0; r < 8; ++r) oacc[t][r] *= alr[r];

    // P A-fragment: S^T C-layout already matches the A layout exactly.
    v16bf pf;
#pragma unroll
    for (int r = 0; r < 8; ++r) { pf[r] = to_bf16(p0[r]); pf[8 + r] = to_bf16(p1[r]); }

#pragma unroll
    for (int t = 0; t < 4; ++t) oacc[t] = wmma_bf16(pf, vf[t], oacc[t]);
  }

  float inv = 1.0f / runS;
  float invr[8];
#pragma unroll
  for (int r = 0; r < 8; ++r) invr[r] = __shfl(inv, r + (hiL ? 8 : 0), 32);
#pragma unroll
  for (int t = 0; t < 4; ++t) {
    int d = h * DV + t * 16 + lx;
#pragma unroll
    for (int r = 0; r < 8; ++r) {
      int n = n0 + r + (hiL ? 8 : 0);
      float val = fmaxf(oacc[t][r] * invr[r], 0.0f);   // ReLU fused
      xx[((long)b * N_ + n) * DH + d] = to_bf16(val);
    }
  }
}

// ---------------------------------------------------------------------------
// Kernel 4: output projection  out[b,o,n] = (wp[o,:] . xx[b,n,:]) * sp + bp
// One wave: 32(o) x 64(n) tile, K = 512 in steps of 32, 8 WMMA per step.
// ---------------------------------------------------------------------------
__global__ void out_gemm(const __bf16* __restrict__ wpb, const __bf16* __restrict__ xx,
                         const float* __restrict__ sp, const float* __restrict__ bp,
                         float* __restrict__ out) {
  int gw   = (int)((blockIdx.x * blockDim.x + threadIdx.x) >> 5);
  int lane = threadIdx.x & 31;
  int lx   = lane & 15;
  bool hi  = lane >= 16;
  // waves: b(8) * o_group(12, 32 channels each) * n_group(16, 64 cols each)
  int ng = gw & 15;
  int rest = gw >> 4;
  int og = rest % 12;
  int b  = rest / 12;
  int o0 = og * 32;
  int nbase = ng * 64;

  const __bf16* wrow0 = wpb + (long)(o0 + lx) * DH;
  const __bf16* wrow1 = wpb + (long)(o0 + 16 + lx) * DH;
  const __bf16* xrow  = xx + ((long)b * N_ + nbase + lx) * DH;

  v8f acc[2][4] = {{{}, {}, {}, {}}, {{}, {}, {}, {}}};
  for (int c0 = 0; c0 < DH; c0 += 32) {
    int aoff = c0 + (hi ? 8 : 0);
    int boff = c0 + (hi ? 8 : 0);
    v16bf af0 = cat16(ld8(wrow0 + aoff), ld8(wrow0 + aoff + 16));
    v16bf af1 = cat16(ld8(wrow1 + aoff), ld8(wrow1 + aoff + 16));
    v16bf bfrag[4];
#pragma unroll
    for (int t = 0; t < 4; ++t) {
      const __bf16* xr = xrow + (long)t * 16 * DH;
      bfrag[t] = cat16(ld8(xr + boff), ld8(xr + boff + 16));
    }
#pragma unroll
    for (int t = 0; t < 4; ++t) {
      acc[0][t] = wmma_bf16(af0, bfrag[t], acc[0][t]);
      acc[1][t] = wmma_bf16(af1, bfrag[t], acc[1][t]);
    }
  }

#pragma unroll
  for (int u = 0; u < 2; ++u) {
#pragma unroll
    for (int t = 0; t < 4; ++t) {
      int n = nbase + t * 16 + lx;
#pragma unroll
      for (int r = 0; r < 8; ++r) {
        int o = o0 + u * 16 + r + (hi ? 8 : 0);
        out[((long)b * C_IN + o) * N_ + n] = acc[u][t][r] * sp[o] + bp[o];
      }
    }
  }
}

// ---------------------------------------------------------------------------
extern "C" void kernel_launch(void* const* d_in, const int* in_sizes, int n_in,
                              void* d_out, int out_size, void* d_ws, size_t ws_size,
                              hipStream_t stream) {
  const float* x  = (const float*)d_in[0];
  const float* wq = (const float*)d_in[1];
  const float* sq = (const float*)d_in[2];
  const float* bq = (const float*)d_in[3];
  const float* wk = (const float*)d_in[4];
  const float* sk = (const float*)d_in[5];
  const float* bk = (const float*)d_in[6];
  const float* wv = (const float*)d_in[7];
  const float* sv = (const float*)d_in[8];
  const float* bv = (const float*)d_in[9];
  const float* wp = (const float*)d_in[10];
  const float* sp = (const float*)d_in[11];
  const float* bp = (const float*)d_in[12];
  float* out = (float*)d_out;

  char* ws = (char*)d_ws;
  size_t off = 0;
  auto carve = [&](size_t bytes) -> void* {
    void* p = ws + off;
    off = (off + bytes + 255) & ~(size_t)255;
    return p;
  };
  __bf16* xT   = (__bf16*)carve((size_t)B_ * N_ * C_IN * 2);      // 6.3 MB
  __bf16* wqkv = (__bf16*)carve((size_t)OQKV * C_IN * 2);         // 0.6 MB
  __bf16* wpb  = (__bf16*)carve((size_t)C_IN * DH * 2);           // 0.4 MB
  __bf16* qo   = (__bf16*)carve((size_t)B_ * NH * N_ * KD * 2);   // 2 MB
  __bf16* ko   = (__bf16*)carve((size_t)B_ * NH * N_ * KD * 2);   // 2 MB
  __bf16* vo   = (__bf16*)carve((size_t)B_ * NH * DV * N_ * 2);   // 8.4 MB
  __bf16* xxb  = (__bf16*)carve((size_t)B_ * N_ * DH * 2);        // 8.4 MB

  // 1) pack: 3145728 + 294912 + 196608 = 3637248 elements
  prep_pack<<<14208, 256, 0, stream>>>(x, wq, wk, wv, wp, xT, wqkv, wpb);

  // 2) QKV projection: 8*24*16 = 3072 waves (32x64 tiles)
  qkv_gemm<<<384, 256, 0, stream>>>(wqkv, xT, sq, bq, sk, bk, sv, bv, qo, ko, vo);

  // 3) fused attention: 8*8*64 = 4096 waves
  attn_fused<<<512, 256, 0, stream>>>(qo, ko, vo, xxb);

  // 4) output projection: 8*12*16 = 1536 waves (32x64 tiles)
  out_gemm<<<192, 256, 0, stream>>>(wpb, xxb, sp, bp, out);
}
